// LTransformerEncoder_35639638622252
// MI455X (gfx1250) — matrix operand order repaired and checked by reading, not compiled, not run on hardware
//
#include <hip/hip_runtime.h>
#include <math.h>

// ---------------------------------------------------------------------------
// LTransformer encoder for MI455X (gfx1250, wave32, WMMA).
// - All dense GEMMs: bf16 WMMA (v_wmma_f32_16x16x32_bf16), fp32 accumulate,
//   128x128x32 tiles, double-buffered LDS.
// - GEMM tile staging uses GLOBAL_LOAD_ASYNC_TO_LDS_B128 (ASYNCcnt-tracked
//   DMA into LDS) -- builtin confirmed present on this toolchain.
// - Attention: WMMA flash-attention, 16 queries/wave, 32-key blocks,
//   online softmax, P re-layout through per-wave LDS.
// ---------------------------------------------------------------------------

typedef __attribute__((ext_vector_type(8)))  float        v8f;
typedef __attribute__((ext_vector_type(16))) __bf16       v16bf;
typedef __attribute__((ext_vector_type(8)))  __bf16       v8bf;
typedef __attribute__((ext_vector_type(4)))  unsigned int u32x4;
typedef __attribute__((ext_vector_type(4)))  int          v4i;

#if defined(__has_builtin)
#if __has_builtin(__builtin_amdgcn_global_load_async_to_lds_b128)
#define ASYNC_LDS 1
#endif
#endif
#ifndef ASYNC_LDS
#define ASYNC_LDS 0
#endif

#if ASYNC_LDS
typedef __attribute__((address_space(1))) v4i gv4i;  // global <4 x i32>
typedef __attribute__((address_space(3))) v4i lv4i;  // LDS    <4 x i32>
#endif

namespace cfg {
constexpr int L  = 12;
constexpr int D  = 512;
constexpr int H  = 8;
constexpr int Bb = 2;
constexpr int S  = 1024;
constexpr int DH = 64;
constexpr int M  = Bb * S;          // 2048 rows
constexpr float KC       = 1.0f;
constexpr float LN_EPS   = 1e-5f;
constexpr float PROJ_EPS = 1e-6f;
}

// ------------------------------- helpers -----------------------------------

__device__ inline float wave_sum(float v) {
#pragma unroll
  for (int o = 16; o > 0; o >>= 1) v += __shfl_xor(v, o, 32);
  return v;
}
__device__ inline float half_max(float v) {
#pragma unroll
  for (int o = 8; o > 0; o >>= 1) v = fmaxf(v, __shfl_xor(v, o, 32));
  return v;
}
__device__ inline float half_sum(float v) {
#pragma unroll
  for (int o = 8; o > 0; o >>= 1) v += __shfl_xor(v, o, 32);
  return v;
}

__device__ inline float gelu_tanh(float x) {
  const float c0 = 0.7978845608028654f, c1 = 0.044715f;
  float u = c0 * (x + c1 * x * x * x);
  return 0.5f * x * (1.0f + tanhf(u));
}

__device__ inline v16bf mk16(v8bf lo, v8bf hi) {
  v16bf a;
#pragma unroll
  for (int e = 0; e < 8; ++e) { a[e] = lo[e]; a[e + 8] = hi[e]; }
  return a;
}

// 16-byte global -> LDS copy; async DMA when available.
__device__ inline void cp16(const __bf16* __restrict__ g, __bf16* s) {
#if ASYNC_LDS
  __builtin_amdgcn_global_load_async_to_lds_b128(
      (gv4i*)(v4i*)const_cast<__bf16*>(g), (lv4i*)(v4i*)s, 0, 0);
#else
  *(u32x4*)s = *(const u32x4*)g;
#endif
}

__device__ inline void wait_async() {
#if ASYNC_LDS
#if defined(__has_builtin) && __has_builtin(__builtin_amdgcn_s_wait_asynccnt)
  __builtin_amdgcn_s_wait_asynccnt(0);
#else
  asm volatile("s_wait_asynccnt 0" ::: "memory");
#endif
#endif
}

// ------------------------- weight convert (transpose) ----------------------
// in:  fp32 [Kd][Nd] row-major   out: bf16 [Np][Kp] row-major (B^T, padded 0)
__global__ void k_convertT(const float* __restrict__ in, int Kd, int Nd,
                           __bf16* __restrict__ out, int Kp, int Np) {
  size_t id  = (size_t)blockIdx.x * 256 + threadIdx.x;
  size_t tot = (size_t)Kp * Np;
  if (id >= tot) return;
  int k = (int)(id % Kp);
  int n = (int)(id / Kp);
  float v = (k < Kd && n < Nd) ? in[(size_t)k * Nd + n] : 0.0f;
  out[id] = (__bf16)v;
}

__global__ void k_f32_to_bf16(const float* __restrict__ in,
                              __bf16* __restrict__ out, int n) {
  int id = blockIdx.x * 256 + threadIdx.x;
  if (id < n) out[id] = (__bf16)in[id];
}

// ------------------------------- embedding ---------------------------------
__global__ void k_embed(const int* __restrict__ toks,
                        const float* __restrict__ table,
                        float* __restrict__ x) {
  int row  = blockIdx.x;
  int lane = threadIdx.x;
  int tok  = toks[row];
  const float* e = table + (size_t)tok * 511;
  float vals[16];
  float ss = 0.0f;
#pragma unroll
  for (int i = 0; i < 16; ++i) {
    int j   = lane + i * 32;
    float v = (j < 511) ? e[j] : 0.0f;
    vals[i] = v;
    ss += v * v;
  }
  ss = wave_sum(ss);
  float* xr = x + (size_t)row * 512;
  if (lane == 0) xr[0] = sqrtf(ss + cfg::KC);
#pragma unroll
  for (int i = 0; i < 16; ++i) {
    int j = lane + i * 32;
    if (j < 511) xr[1 + j] = vals[i];
  }
}

// ------------------------------- layernorm ---------------------------------
__global__ void k_layernorm(const float* __restrict__ in, int ld, int coff,
                            const float* __restrict__ g,
                            const float* __restrict__ b,
                            __bf16* __restrict__ out_bf,
                            float* __restrict__ out_f) {
  int row  = blockIdx.x;
  int lane = threadIdx.x;
  const float* r = in + (size_t)row * ld + coff;
  float vals[16];
  float sum = 0.0f;
#pragma unroll
  for (int i = 0; i < 16; ++i) {
    int j   = lane + i * 32;
    float v = (j < 511) ? r[j] : 0.0f;
    vals[i] = v;
    sum += v;
  }
  sum = wave_sum(sum);
  float mu = sum * (1.0f / 511.0f);
  float var = 0.0f;
#pragma unroll
  for (int i = 0; i < 16; ++i) {
    int j = lane + i * 32;
    if (j < 511) { float d = vals[i] - mu; var += d * d; }
  }
  var = wave_sum(var) * (1.0f / 511.0f);
  float rs = rsqrtf(var + cfg::LN_EPS);
  float ss = 0.0f;
#pragma unroll
  for (int i = 0; i < 16; ++i) {
    int j = lane + i * 32;
    if (j < 511) {
      float y = (vals[i] - mu) * rs * g[j] + b[j];
      vals[i] = y;
      ss += y * y;
    }
  }
  ss = wave_sum(ss);
  float t = sqrtf(ss + cfg::KC);
  if (out_bf) {
    __bf16* o = out_bf + (size_t)row * 512;
    if (lane == 0) o[0] = (__bf16)t;
#pragma unroll
    for (int i = 0; i < 16; ++i) {
      int j = lane + i * 32;
      if (j < 511) o[1 + j] = (__bf16)vals[i];
    }
  } else {
    float* o = out_f + (size_t)row * 512;
    if (lane == 0) o[0] = t;
#pragma unroll
    for (int i = 0; i < 16; ++i) {
      int j = lane + i * 32;
      if (j < 511) o[1 + j] = vals[i];
    }
  }
}

// ------------------------------ WMMA GEMM ----------------------------------
__device__ inline void load_tile(const __bf16* __restrict__ g, int row0,
                                 int ldg, int kk, __bf16* s, int tid) {
#pragma unroll
  for (int i = 0; i < 2; ++i) {
    int ch = tid + i * 256;          // 512 chunks of 8 bf16
    int r  = ch >> 2;
    int c  = (ch & 3) * 8;
    cp16(g + (size_t)(row0 + r) * ldg + kk + c, s + r * 32 + c);
  }
}

__global__ void __launch_bounds__(256)
k_gemm_bf16(const __bf16* __restrict__ A, const __bf16* __restrict__ BT,
            const float* __restrict__ bias, float* __restrict__ C,
            int Kpad, int Nreal, int ldc, int act) {
  __shared__ __align__(16) __bf16 sA[2][128 * 32];
  __shared__ __align__(16) __bf16 sB[2][128 * 32];

  const int tid   = threadIdx.x;
  const int m0    = blockIdx.y * 128;
  const int n0    = blockIdx.x * 128;
  const int lane  = tid & 31;
  const int wave  = tid >> 5;
  const int wm    = wave & 3;   // 4 waves along M
  const int wn    = wave >> 2;  // 2 waves along N
  const int lmod  = lane & 15;
  const int lhalf = lane >> 4;

  v8f acc[2][4];
#pragma unroll
  for (int mt = 0; mt < 2; ++mt)
#pragma unroll
    for (int nt = 0; nt < 4; ++nt)
#pragma unroll
      for (int e = 0; e < 8; ++e) acc[mt][nt][e] = 0.0f;

  const int KT = Kpad / 32;

  load_tile(A, m0, Kpad, 0, sA[0], tid);
  load_tile(BT, n0, Kpad, 0, sB[0], tid);
  wait_async();
  __syncthreads();

  for (int kt = 0; kt < KT; ++kt) {
    const int cur = kt & 1, nxt = cur ^ 1;
    if (kt + 1 < KT) {
      load_tile(A, m0, Kpad, (kt + 1) * 32, sA[nxt], tid);
      load_tile(BT, n0, Kpad, (kt + 1) * 32, sB[nxt], tid);
    }
    if (kt + 2 < KT) {  // L2 prefetch two tiles ahead
      __builtin_prefetch(A + (size_t)(m0 + (tid >> 1)) * Kpad + (kt + 2) * 32,
                         0, 1);
      __builtin_prefetch(BT + (size_t)(n0 + (tid >> 1)) * Kpad + (kt + 2) * 32,
                         0, 1);
    }

    v16bf af[2], bfg[4];
#pragma unroll
    for (int mt = 0; mt < 2; ++mt) {
      const __bf16* p = &sA[cur][(wm * 32 + mt * 16 + lmod) * 32];
      af[mt] = mk16(*(const v8bf*)(p + lhalf * 8),
                    *(const v8bf*)(p + 16 + lhalf * 8));
    }
#pragma unroll
    for (int nt = 0; nt < 4; ++nt) {
      const __bf16* p = &sB[cur][(wn * 64 + nt * 16 + lmod) * 32];
      bfg[nt] = mk16(*(const v8bf*)(p + lhalf * 8), *(const v8bf*)(p + 16 + lhalf * 8));
    }

#pragma unroll
    for (int mt = 0; mt < 2; ++mt)
#pragma unroll
      for (int nt = 0; nt < 4; ++nt)
        acc[mt][nt] = __builtin_amdgcn_wmma_f32_16x16x32_bf16(
            false, af[mt], false, bfg[nt], (short)0, acc[mt][nt], false, false);

    wait_async();
    __syncthreads();
  }

#pragma unroll
  for (int mt = 0; mt < 2; ++mt) {
#pragma unroll
    for (int nt = 0; nt < 4; ++nt) {
#pragma unroll
      for (int r = 0; r < 8; ++r) {
        int row = m0 + wm * 32 + mt * 16 + lhalf * 8 + r;
        int col = n0 + wn * 64 + nt * 16 + lmod;
        if (col < Nreal) {
          float v = acc[mt][nt][r] + bias[col];
          if (act) v = gelu_tanh(v);
          C[(size_t)row * ldc + col] = v;
        }
      }
    }
  }
}

// -------------- per-head add_time + bf16 pack for WMMA attention -----------
__global__ void k_head_pack(const float* __restrict__ qlin,
                            const float* __restrict__ klin,
                            const float* __restrict__ vlin,
                            __bf16* __restrict__ qn, __bf16* __restrict__ kb,
                            __bf16* __restrict__ vt) {
  int id   = blockIdx.x;  // row*H + h
  int h    = id % cfg::H;
  int row  = id / cfg::H;
  int b    = row / cfg::S;
  int s    = row % cfg::S;
  int bh   = b * cfg::H + h;
  int lane = threadIdx.x;
  size_t base = (size_t)row * 512 + h * 64;

  float q0 = qlin[base + lane], q1 = qlin[base + 32 + lane];
  float k0 = klin[base + lane], k1 = klin[base + 32 + lane];
  float v0 = vlin[base + lane], v1 = vlin[base + 32 + lane];
  float tq = sqrtf(wave_sum(q0 * q0 + q1 * q1) + cfg::KC);
  float tk = sqrtf(wave_sum(k0 * k0 + k1 * k1) + cfg::KC);
  float tv = sqrtf(wave_sum(v0 * v0 + v1 * v1) + cfg::KC);

  __bf16* oq = qn + ((size_t)bh * cfg::S + s) * 96;
  __bf16* ok = kb + ((size_t)bh * cfg::S + s) * 96;
  if (lane == 0) { oq[0] = (__bf16)(-tq); ok[0] = (__bf16)tk; }
  oq[1 + lane]  = (__bf16)q0;
  oq[33 + lane] = (__bf16)q1;
  ok[1 + lane]  = (__bf16)k0;
  ok[33 + lane] = (__bf16)k1;
  if (lane < 31) { oq[65 + lane] = (__bf16)0.0f; ok[65 + lane] = (__bf16)0.0f; }

  __bf16* ov = vt + (size_t)bh * 96 * cfg::S;
  if (lane == 0) ov[s] = (__bf16)tv;
  ov[(size_t)(1 + lane) * cfg::S + s]  = (__bf16)v0;
  ov[(size_t)(33 + lane) * cfg::S + s] = (__bf16)v1;
  if (lane < 31) ov[(size_t)(65 + lane) * cfg::S + s] = (__bf16)0.0f;
}

// ------------------------- WMMA flash attention ----------------------------
__global__ void __launch_bounds__(256)
k_attn_wmma(const __bf16* __restrict__ qn, const __bf16* __restrict__ kb,
            const __bf16* __restrict__ vt, float* __restrict__ aggh) {
  __shared__ __align__(16) __bf16 sP[8][16 * 32];  // per-wave P tile

  const int tid   = threadIdx.x;
  const int lane  = tid & 31;
  const int wave  = tid >> 5;
  const int lmod  = lane & 15;
  const int lhalf = lane >> 4;
  const int bh    = blockIdx.y;
  const int q0    = blockIdx.x * 128 + wave * 16;

  const __bf16* qbase = qn + ((size_t)bh * cfg::S + q0) * 96;
  const __bf16* kbase = kb + (size_t)bh * cfg::S * 96;
  const __bf16* vbase = vt + (size_t)bh * 96 * cfg::S;

  v16bf qf[3];
#pragma unroll
  for (int c = 0; c < 3; ++c) {
    const __bf16* p = qbase + (size_t)lmod * 96 + c * 32;
    qf[c] = mk16(*(const v8bf*)(p + lhalf * 8),
                 *(const v8bf*)(p + 16 + lhalf * 8));
  }

  v8f O[6];
#pragma unroll
  for (int d = 0; d < 6; ++d)
#pragma unroll
    for (int e = 0; e < 8; ++e) O[d][e] = 0.0f;
  float mrun[8], lrun[8];
#pragma unroll
  for (int r = 0; r < 8; ++r) { mrun[r] = -3.0e38f; lrun[r] = 0.0f; }

  const int nk = q0 + 16;  // causal: keys < nk
  for (int jb = 0; jb < nk; jb += 32) {
    v8f c0, c1;
#pragma unroll
    for (int e = 0; e < 8; ++e) { c0[e] = 0.0f; c1[e] = 0.0f; }
#pragma unroll
    for (int c = 0; c < 3; ++c) {
      int kr0 = jb + lmod;       if (kr0 >= cfg::S) kr0 = cfg::S - 1;
      int kr1 = jb + 16 + lmod;  if (kr1 >= cfg::S) kr1 = cfg::S - 1;
      const __bf16* p0 = kbase + (size_t)kr0 * 96 + c * 32 + lhalf * 8;
      const __bf16* p1 = kbase + (size_t)kr1 * 96 + c * 32 + lhalf * 8;
      v16bf kf0 = mk16(*(const v8bf*)p0, *(const v8bf*)(p0 + 16));
      v16bf kf1 = mk16(*(const v8bf*)p1, *(const v8bf*)(p1 + 16));
      c0 = __builtin_amdgcn_wmma_f32_16x16x32_bf16(false, qf[c], false, kf0,
                                                   (short)0, c0, false, false);
      c1 = __builtin_amdgcn_wmma_f32_16x16x32_bf16(false, qf[c], false, kf1,
                                                   (short)0, c1, false, false);
    }

    float p0v[8], p1v[8];
#pragma unroll
    for (int r = 0; r < 8; ++r) {
      int qi = q0 + lhalf * 8 + r;
      int k0 = jb + lmod, k1 = jb + 16 + lmod;
      float s0 = 0.25f + 0.25f * c0[r];  // (2K + 2*linner) / sqrt(DH)
      float s1 = 0.25f + 0.25f * c1[r];
      if (k0 > qi) s0 = -1.0e9f;
      if (k1 > qi) s1 = -1.0e9f;
      float rm   = half_max(fmaxf(s0, s1));
      float mn   = fmaxf(mrun[r], rm);
      float corr = __expf(mrun[r] - mn);
      p0v[r] = __expf(s0 - mn);
      p1v[r] = __expf(s1 - mn);
      float rs = half_sum(p0v[r] + p1v[r]);
      lrun[r] = lrun[r] * corr + rs;
      mrun[r] = mn;
#pragma unroll
      for (int d = 0; d < 6; ++d) O[d][r] *= corr;
    }

    __bf16* sp = sP[wave];
#pragma unroll
    for (int r = 0; r < 8; ++r) {
      int row = lhalf * 8 + r;
      sp[row * 32 + lmod]      = (__bf16)p0v[r];
      sp[row * 32 + 16 + lmod] = (__bf16)p1v[r];
    }
    const __bf16* pp = sp + lmod * 32;
    v16bf pf = mk16(*(const v8bf*)(pp + lhalf * 8),
                    *(const v8bf*)(pp + 16 + lhalf * 8));

#pragma unroll
    for (int d = 0; d < 6; ++d) {
      const __bf16* p =
          vbase + (size_t)(d * 16 + lmod) * cfg::S + jb + lhalf * 8;
      v16bf vf = mk16(*(const v8bf*)p, *(const v8bf*)(p + 16));
      O[d] = __builtin_amdgcn_wmma_f32_16x16x32_bf16(false, pf, false, vf,
                                                     (short)0, O[d], false,
                                                     false);
    }
  }

#pragma unroll
  for (int d = 0; d < 6; ++d) {
#pragma unroll
    for (int r = 0; r < 8; ++r) {
      int dim = d * 16 + lmod;
      if (dim < 65) {
        int qi = q0 + lhalf * 8 + r;
        aggh[((size_t)bh * cfg::S + qi) * 65 + dim] = O[d][r] / lrun[r];
      }
    }
  }
}

// --------------- l_project(agg) + pack heads to bf16 [M][544] --------------
__global__ void k_lproj_pack(const float* __restrict__ aggh,
                             __bf16* __restrict__ ao) {
  int id   = blockIdx.x;  // row*H + h
  int h    = id % cfg::H;
  int row  = id / cfg::H;
  int b    = row / cfg::S;
  int s    = row % cfg::S;
  int lane = threadIdx.x;
  const float* z = aggh + (((size_t)(b * cfg::H + h)) * cfg::S + s) * 65;
  float zv[3];
  float tot = 0.0f;
#pragma unroll
  for (int i = 0; i < 3; ++i) {
    int j   = lane + i * 32;
    float v = (j < 65) ? z[j] : 0.0f;
    zv[i]   = v;
    tot += v * v;
  }
  tot = wave_sum(tot);
  float z0 = __shfl(zv[0], 0, 32);
  float f = rsqrtf(fmaxf(2.0f * z0 * z0 - tot, cfg::PROJ_EPS));
  __bf16* o = ao + (size_t)row * 544 + h * 65;
#pragma unroll
  for (int i = 0; i < 3; ++i) {
    int j = lane + i * 32;
    if (j < 65) o[j] = (__bf16)(zv[i] * f);
  }
  if (h == cfg::H - 1 && lane < 24)
    ao[(size_t)row * 544 + 520 + lane] = (__bf16)0.0f;  // zero K-pad
}

// ----------------- x = l_project(x + w * add_time(delta)) ------------------
__global__ void k_resid_lproj(float* __restrict__ x,
                              const float* __restrict__ delta, int ldd,
                              const float* __restrict__ wv, int widx) {
  int row  = blockIdx.x;
  int lane = threadIdx.x;
  float w  = wv[widx];
  const float* dr = delta + (size_t)row * ldd;
  float ss = 0.0f;
#pragma unroll
  for (int i = 0; i < 16; ++i) {
    int j = lane + i * 32;
    if (j < 511) { float v = dr[j]; ss += v * v; }
  }
  ss = wave_sum(ss);
  float t = sqrtf(ss + cfg::KC);
  float* xr = x + (size_t)row * 512;
  float y[16];
  float tot = 0.0f;
#pragma unroll
  for (int i = 0; i < 16; ++i) {
    int j    = lane + i * 32;
    float a  = (j == 0) ? t : dr[j - 1];
    float yv = xr[j] + w * a;
    y[i] = yv;
    tot += yv * yv;
  }
  tot = wave_sum(tot);
  float y0 = __shfl(y[0], 0, 32);
  float f  = rsqrtf(fmaxf(2.0f * y0 * y0 - tot, cfg::PROJ_EPS));
#pragma unroll
  for (int i = 0; i < 16; ++i) {
    int j = lane + i * 32;
    xr[j] = y[i] * f;
  }
}

// ------------------ add_time over n fp32 cols -> bf16 [1+n] ----------------
__global__ void k_addtime_bf16(const float* __restrict__ in, int ld, int n,
                               __bf16* __restrict__ out, int ldo) {
  int row  = blockIdx.x;
  int lane = threadIdx.x;
  const float* r = in + (size_t)row * ld;
  float ss = 0.0f;
  for (int j = lane; j < n; j += 32) { float v = r[j]; ss += v * v; }
  ss = wave_sum(ss);
  __bf16* o = out + (size_t)row * ldo;
  if (lane == 0) o[0] = (__bf16)sqrtf(ss + cfg::KC);
  for (int j = lane; j < n; j += 32) o[1 + j] = (__bf16)r[j];
}

// ---------------------------------------------------------------------------
extern "C" void kernel_launch(void* const* d_in, const int* in_sizes, int n_in,
                              void* d_out, int out_size, void* d_ws,
                              size_t ws_size, hipStream_t stream) {
  (void)in_sizes; (void)n_in; (void)out_size;
  using namespace cfg;

  const int*   toks  = (const int*)d_in[0];
  const float* embed = (const float*)d_in[1];
  const float* Wq    = (const float*)d_in[2];
  const float* bq    = (const float*)d_in[3];
  const float* Wk    = (const float*)d_in[4];
  const float* bk    = (const float*)d_in[5];
  const float* Wv    = (const float*)d_in[6];
  const float* bv    = (const float*)d_in[7];
  const float* Wo    = (const float*)d_in[8];
  const float* bo    = (const float*)d_in[9];
  const float* ln1g  = (const float*)d_in[10];
  const float* ln1b  = (const float*)d_in[11];
  const float* ln2g  = (const float*)d_in[12];
  const float* ln2b  = (const float*)d_in[13];
  const float* Wfc   = (const float*)d_in[14];
  const float* bfc   = (const float*)d_in[15];
  const float* Wpr   = (const float*)d_in[16];
  const float* bpr   = (const float*)d_in[17];
  const float* rw1   = (const float*)d_in[18];
  const float* rw2   = (const float*)d_in[19];
  const float* Wfin  = (const float*)d_in[20];
  const float* bfin  = (const float*)d_in[21];
  const float* lnfg  = (const float*)d_in[22];
  const float* lnfb  = (const float*)d_in[23];

  char*  base = (char*)d_ws;
  size_t off  = 0;
  auto alloc = [&](size_t bytes) -> void* {
    void* p = base + off;
    off = (off + bytes + 255) & ~(size_t)255;
    return p;
  };

  __bf16* wqT   = (__bf16*)alloc((size_t)L * 512 * 512 * 2);
  __bf16* wkT   = (__bf16*)alloc((size_t)L * 512 * 512 * 2);
  __bf16* wvT   = (__bf16*)alloc((size_t)L * 512 * 512 * 2);
  __bf16* woT   = (__bf16*)alloc((size_t)L * 512 * 544 * 2);
  __bf16* wfcT  = (__bf16*)alloc((size_t)L * 2048 * 512 * 2);
  __bf16* wprT  = (__bf16*)alloc((size_t)L * 512 * 2048 * 2);
  __bf16* wfinT = (__bf16*)alloc((size_t)512 * 512 * 2);

  float*  x      = (float*)alloc((size_t)M * 512 * 4);
  __bf16* x_bf   = (__bf16*)alloc((size_t)M * 512 * 2);
  __bf16* lx_bf  = (__bf16*)alloc((size_t)M * 512 * 2);
  float*  qlin   = (float*)alloc((size_t)M * 512 * 4);
  float*  klin   = (float*)alloc((size_t)M * 512 * 4);
  float*  vlin   = (float*)alloc((size_t)M * 512 * 4);
  __bf16* qn_bf  = (__bf16*)alloc((size_t)Bb * H * S * 96 * 2);
  __bf16* kh_bf  = (__bf16*)alloc((size_t)Bb * H * S * 96 * 2);
  __bf16* vt_bf  = (__bf16*)alloc((size_t)Bb * H * 96 * S * 2);
  float*  aggh   = (float*)alloc((size_t)Bb * H * S * 65 * 4);
  __bf16* ao_bf  = (__bf16*)alloc((size_t)M * 544 * 2);
  float*  wo_out = (float*)alloc((size_t)M * 512 * 4);
  float*  fc_out = (float*)alloc((size_t)M * 2048 * 4);
  __bf16* h2_bf  = (__bf16*)alloc((size_t)M * 2048 * 2);
  float*  pr_out = (float*)alloc((size_t)M * 512 * 4);
  float*  fin_out= (float*)alloc((size_t)M * 512 * 4);

  if (off > ws_size) return;  // workspace too small; bail deterministically

  auto convT = [&](const float* in, int Kd, int Nd, __bf16* out, int Kp,
                   int Np) {
    size_t tot = (size_t)Kp * Np;
    k_convertT<<<(unsigned)((tot + 255) / 256), 256, 0, stream>>>(in, Kd, Nd,
                                                                  out, Kp, Np);
  };
  auto gemm = [&](const __bf16* Ap, const __bf16* Bp, const float* bias,
                  float* Cp, int Kp, int Np, int Nreal, int ldc, int act) {
    dim3 g(Np / 128, M / 128);
    k_gemm_bf16<<<g, 256, 0, stream>>>(Ap, Bp, bias, Cp, Kp, Nreal, ldc, act);
  };

  for (int i = 0; i < L; ++i) {
    convT(Wq + (size_t)i * D * D, D, D, wqT + (size_t)i * 512 * 512, 512, 512);
    convT(Wk + (size_t)i * D * D, D, D, wkT + (size_t)i * 512 * 512, 512, 512);
    convT(Wv + (size_t)i * D * D, D, D, wvT + (size_t)i * 512 * 512, 512, 512);
    convT(Wo + (size_t)i * 520 * 511, 520, 511,
          woT + (size_t)i * 512 * 544, 544, 512);
    convT(Wfc + (size_t)i * 512 * 2047, 512, 2047,
          wfcT + (size_t)i * 2048 * 512, 512, 2048);
    convT(Wpr + (size_t)i * 2048 * 511, 2048, 511,
          wprT + (size_t)i * 512 * 2048, 2048, 512);
  }
  convT(Wfin, 512, 511, wfinT, 512, 512);

  k_embed<<<M, 32, 0, stream>>>(toks, embed, x);

  for (int i = 0; i < L; ++i) {
    k_layernorm<<<M, 32, 0, stream>>>(x, 512, 1, ln1g + (size_t)i * 511,
                                      ln1b + (size_t)i * 511, lx_bf, nullptr);

    gemm(lx_bf, wqT + (size_t)i * 512 * 512, bq + (size_t)i * 512, qlin, 512,
         512, 512, 512, 0);
    gemm(lx_bf, wkT + (size_t)i * 512 * 512, bk + (size_t)i * 512, klin, 512,
         512, 512, 512, 0);
    gemm(lx_bf, wvT + (size_t)i * 512 * 512, bv + (size_t)i * 512, vlin, 512,
         512, 512, 512, 0);

    k_head_pack<<<M * H, 32, 0, stream>>>(qlin, klin, vlin, qn_bf, kh_bf,
                                          vt_bf);
    {
      dim3 ag(S / 128, Bb * H);
      k_attn_wmma<<<ag, 256, 0, stream>>>(qn_bf, kh_bf, vt_bf, aggh);
    }
    k_lproj_pack<<<M * H, 32, 0, stream>>>(aggh, ao_bf);

    gemm(ao_bf, woT + (size_t)i * 512 * 544, bo + (size_t)i * 511, wo_out, 544,
         512, 511, 512, 0);
    k_resid_lproj<<<M, 32, 0, stream>>>(x, wo_out, 512, rw1, i);

    k_layernorm<<<M, 32, 0, stream>>>(x, 512, 1, ln2g + (size_t)i * 511,
                                      ln2b + (size_t)i * 511, lx_bf, nullptr);
    gemm(lx_bf, wfcT + (size_t)i * 2048 * 512, bfc + (size_t)i * 2047, fc_out,
         512, 2048, 2047, 2048, 1);  // fused gelu
    k_addtime_bf16<<<M, 32, 0, stream>>>(fc_out, 2048, 2047, h2_bf, 2048);
    gemm(h2_bf, wprT + (size_t)i * 512 * 2048, bpr + (size_t)i * 511, pr_out,
         2048, 512, 511, 512, 0);
    k_resid_lproj<<<M, 32, 0, stream>>>(x, pr_out, 512, rw2, i);
  }

  k_f32_to_bf16<<<(M * 512 + 255) / 256, 256, 0, stream>>>(x, x_bf, M * 512);
  gemm(x_bf, wfinT, bfin, fin_out, 512, 512, 511, 512, 0);
  k_layernorm<<<M, 32, 0, stream>>>(fin_out, 512, 0, lnfg, lnfb, nullptr,
                                    (float*)d_out);
}
